// LLaMa_MLA_61916248539451
// MI455X (gfx1250) — matrix operand order repaired and checked by reading, not compiled, not run on hardware
//
#include <hip/hip_runtime.h>
#include <hip/hip_bf16.h>
#include <cstdint>
#include <cstddef>

// ---------------------------------------------------------------------------
// Types
// ---------------------------------------------------------------------------
typedef __bf16 bf16_t;
typedef __attribute__((ext_vector_type(16))) __bf16 v16bf;
typedef __attribute__((ext_vector_type(8)))  __bf16 v8bf;
typedef __attribute__((ext_vector_type(8)))  float  v8f;

// ---------------------------------------------------------------------------
// Model dims
// ---------------------------------------------------------------------------
#define BB      2
#define SS      1024
#define DD      1024
#define HH      16
#define HDIM    64
#define CKVD    256
#define CQD     384
#define RRD     32
#define DFFD    2816
#define NLAYERS 4
#define VOCAB   32000
#define MTOK    (BB*SS)          // 2048 token rows
#define DQK     (HDIM+RRD)       // 96
#define EPSF    1.1920929e-07f

// ---------------------------------------------------------------------------
// helpers
// ---------------------------------------------------------------------------
__device__ __forceinline__ v16bf mk16(v8bf lo, v8bf hi) {
  return __builtin_shufflevector(lo, hi, 0,1,2,3,4,5,6,7,8,9,10,11,12,13,14,15);
}
__device__ __forceinline__ v8f wmma_bf16(v16bf a, v16bf b, v8f c) {
  return __builtin_amdgcn_wmma_f32_16x16x32_bf16(false, a, false, b, (short)0, c, false, false);
}

// Async global -> LDS copy, 16B per lane, tracked by ASYNCcnt (gfx1250).
__device__ __forceinline__ void async_load_b128(const void* gsrc, void* lds_dst) {
  unsigned lds = (unsigned)(size_t)lds_dst;                 // low 32 bits = LDS offset
  unsigned long long g = (unsigned long long)(size_t)gsrc;  // 64-bit global address
  asm volatile("global_load_async_to_lds_b128 %0, %1, off"
               :: "v"(lds), "v"(g) : "memory");
}
__device__ __forceinline__ void wait_async0() {
  asm volatile("s_wait_asynccnt 0x0" ::: "memory");
}

// ---------------------------------------------------------------------------
// Embedding gather: h[row, :] = embed[tok[row], :]
// ---------------------------------------------------------------------------
__global__ void embed_kernel(const int* __restrict__ tokens,
                             const float* __restrict__ embed,
                             float* __restrict__ h)
{
  int row = blockIdx.x;
  int tok = tokens[row];
  const float* src = embed + (size_t)tok * DD;
  float* dst = h + (size_t)row * DD;
  for (int i = threadIdx.x; i < DD; i += blockDim.x) dst[i] = src[i];
}

// ---------------------------------------------------------------------------
// RMSNorm (fp32 in, bf16 out). x rows have stride xstride, out rows compact.
// ---------------------------------------------------------------------------
__global__ void rmsnorm_kernel(const float* __restrict__ x, int xstride,
                               const float* __restrict__ w,
                               bf16_t* __restrict__ out, int ncols)
{
  __shared__ float red[9];
  int row = blockIdx.x;
  const float* xr = x + (size_t)row * xstride;
  float ss = 0.f;
  for (int i = threadIdx.x; i < ncols; i += blockDim.x) { float v = xr[i]; ss += v * v; }
#pragma unroll
  for (int off = 16; off > 0; off >>= 1) ss += __shfl_xor(ss, off, 32);
  int wid = threadIdx.x >> 5;
  if ((threadIdx.x & 31) == 0) red[wid] = ss;
  __syncthreads();
  if (threadIdx.x == 0) {
    float tot = 0.f;
    int nw = blockDim.x >> 5;
    for (int i = 0; i < nw; ++i) tot += red[i];
    red[8] = rsqrtf(tot / (float)ncols + EPSF);
  }
  __syncthreads();
  float sc = red[8];
  bf16_t* o = out + (size_t)row * ncols;
  for (int i = threadIdx.x; i < ncols; i += blockDim.x)
    o[i] = (bf16_t)(xr[i] * sc * w[i]);
}

// ---------------------------------------------------------------------------
// Weight transpose + convert: W[K,N] fp32 -> Wt[N,K] bf16  (N-major for WMMA B)
// ---------------------------------------------------------------------------
__global__ void transpose_bf16_kernel(const float* __restrict__ W,
                                      bf16_t* __restrict__ Wt, int K, int N)
{
  __shared__ float tile[32][33];
  int kb = blockIdx.x * 32, nb = blockIdx.y * 32;
  int tx = threadIdx.x & 31, ty = threadIdx.x >> 5;   // 256 threads: 32x8
  for (int i = ty; i < 32; i += 8) {
    int k = kb + i, n = nb + tx;
    tile[i][tx] = (k < K && n < N) ? W[(size_t)k * N + n] : 0.f;
  }
  __syncthreads();
  for (int i = ty; i < 32; i += 8) {
    int n = nb + i, k = kb + tx;
    if (n < N && k < K) Wt[(size_t)n * K + k] = (bf16_t)tile[tx][i];
  }
}

// ---------------------------------------------------------------------------
// WMMA GEMM: C[M,N] = A[M,K](bf16) * Wt[N,K]^T(bf16) + bias[N] (+ residual)
// Block tile 128x128, BK=32, 256 threads = 8 waves; wave owns 32x64 (8 WMMAs).
// Double-buffered LDS fed by GLOBAL_LOAD_ASYNC_TO_LDS_B128 (ASYNCcnt).
// ---------------------------------------------------------------------------
#define GBM 128
#define GBN 128
#define GBK 32

__device__ __forceinline__ void gemm_store_tile(float* C, const float* bias,
                                                const float* residual, const v8f& acc,
                                                int mbase, int n, int N, int hi)
{
  if (n >= N) return;
  float bv = bias ? bias[n] : 0.f;
#pragma unroll
  for (int r = 0; r < 8; ++r) {
    int m = mbase + hi * 8 + r;
    size_t idx = (size_t)m * N + n;
    float v = acc[r] + bv;
    if (residual) v += residual[idx];
    C[idx] = v;
  }
}

__global__ __launch_bounds__(256)
void gemm_bf16_kernel(const bf16_t* __restrict__ A,   // M x K
                      const bf16_t* __restrict__ Wt,  // N x K
                      const float* __restrict__ bias, // N (or null)
                      const float* residual,          // M x N (or null)
                      float* C, int M, int N, int K)
{
  __shared__ bf16_t lA[2][GBM * GBK];
  __shared__ bf16_t lB[2][GBM * GBK];

  int m0 = blockIdx.y * GBM;
  int n0 = blockIdx.x * GBN;
  int t = threadIdx.x;
  int lane = t & 31, w = t >> 5;
  int wm = w & 3, wn = w >> 2;          // wave tile: rows wm*32, cols wn*64
  int hi = lane >> 4, l16 = lane & 15;

  // staging chunks: thread t covers rows r0 and r0+64, 8-half column c0
  int r0 = t >> 2;                      // 0..63
  int c0 = (t & 3) * 8;                 // 0,8,16,24

  // Pre-zero LDS rows of B that are out of range in N; async copies never
  // touch them, so the zeros persist across all K-steps.
  if (n0 + GBM > N) {
    for (int i = t; i < GBM * 4; i += 256) {
      int row = i >> 2, col = (i & 3) * 8;
      if (n0 + row >= N) {
        v8bf z = {};
        *(v8bf*)&lB[0][row * GBK + col] = z;
        *(v8bf*)&lB[1][row * GBK + col] = z;
      }
    }
  }

  v8f acc[2][4] = {};

  auto stage = [&](int buf, int k0) {
    async_load_b128(A + (size_t)(m0 + r0) * K + k0 + c0,        &lA[buf][r0 * GBK + c0]);
    async_load_b128(A + (size_t)(m0 + r0 + 64) * K + k0 + c0,   &lA[buf][(r0 + 64) * GBK + c0]);
    if (n0 + r0 < N)
      async_load_b128(Wt + (size_t)(n0 + r0) * K + k0 + c0,     &lB[buf][r0 * GBK + c0]);
    if (n0 + r0 + 64 < N)
      async_load_b128(Wt + (size_t)(n0 + r0 + 64) * K + k0 + c0,&lB[buf][(r0 + 64) * GBK + c0]);
  };

  stage(0, 0);
  wait_async0();
  __syncthreads();

  int buf = 0;
  for (int k0 = 0; k0 < K; k0 += GBK) {
    if (k0 + GBK < K) stage(buf ^ 1, k0 + GBK);   // overlap next tile with compute

    // Preload ALL fragments first: 12x ds_load_b128 stream into the LDS pipe,
    // then 8 WMMAs issue back-to-back with a single dscnt wait.
    v16bf af0, af1;
    {
      int am = wm * 32 + l16;
      af0 = mk16(*(const v8bf*)&lA[buf][am * GBK + hi * 8],
                 *(const v8bf*)&lA[buf][am * GBK + 16 + hi * 8]);
      af1 = mk16(*(const v8bf*)&lA[buf][(am + 16) * GBK + hi * 8],
                 *(const v8bf*)&lA[buf][(am + 16) * GBK + 16 + hi * 8]);
    }
    v16bf bfr[4];
#pragma unroll
    for (int ni = 0; ni < 4; ++ni) {
      int bn = wn * 64 + ni * 16 + l16;
      bfr[ni] = mk16(*(const v8bf*)&lB[buf][bn * GBK + hi * 8],
                     *(const v8bf*)&lB[buf][bn * GBK + 16 + hi * 8]);
    }
#pragma unroll
    for (int ni = 0; ni < 4; ++ni) {
      acc[0][ni] = wmma_bf16(af0, bfr[ni], acc[0][ni]);
      acc[1][ni] = wmma_bf16(af1, bfr[ni], acc[1][ni]);
    }

    wait_async0();        // next tile landed in LDS
    __syncthreads();      // everyone done reading current buffer
    buf ^= 1;
  }

#pragma unroll
  for (int mi = 0; mi < 2; ++mi) {
    int mbase = m0 + wm * 32 + mi * 16;
#pragma unroll
    for (int ni = 0; ni < 4; ++ni)
      gemm_store_tile(C, bias, residual, acc[mi][ni],
                      mbase, n0 + wn * 64 + ni * 16 + l16, N, hi);
  }
}

// ---------------------------------------------------------------------------
// RoPE pack: q_up [B,S,H,96] fp32 -> Qc [B,H,S,96] bf16 with rope on last 32
// ---------------------------------------------------------------------------
__global__ void build_q_kernel(const float* __restrict__ qf, bf16_t* __restrict__ Qc)
{
  int idx = blockIdx.x;                 // (b*H + h)*S + s
  int s = idx % SS, h = (idx / SS) % HH, b = idx / (SS * HH);
  const float* src = qf + ((size_t)(b * SS + s) * HH + h) * DQK;
  bf16_t* dst = Qc + (size_t)idx * DQK;
  int d = threadIdx.x;                  // 0..95
  float v;
  if (d < HDIM) {
    v = src[d];
  } else {
    int j = d - HDIM, jj = j & 15;
    float theta = __powf(10000.0f, -(float)jj / 16.0f);
    float ang = (float)s * theta;
    float c = __cosf(ang), sn = __sinf(ang);
    v = (j < 16) ? (src[HDIM + j] * c - src[HDIM + j + 16] * sn)
                 : (src[HDIM + j] * c + src[HDIM + j - 16] * sn);
  }
  dst[d] = (bf16_t)v;
}

// ---------------------------------------------------------------------------
// K/V pack: kv_up [B,S,H,128] fp32 + ckv [B,S,288] (rope cols 256..287)
//  -> Kc [B,H,S,96] bf16 (k | rope(k_rope) broadcast over heads), Vb [B,H,S,64]
// ---------------------------------------------------------------------------
__global__ void build_kv_kernel(const float* __restrict__ kvup,
                                const float* __restrict__ ckv,
                                bf16_t* __restrict__ Kc, bf16_t* __restrict__ Vb)
{
  int idx = blockIdx.x;
  int s = idx % SS, h = (idx / SS) % HH, b = idx / (SS * HH);
  const float* kvsrc = kvup + ((size_t)(b * SS + s) * HH + h) * (2 * HDIM);
  const float* kr = ckv + (size_t)(b * SS + s) * (CKVD + RRD) + CKVD;
  int d = threadIdx.x;                  // 0..95
  float v;
  if (d < HDIM) {
    v = kvsrc[d];
    Vb[(size_t)idx * HDIM + d] = (bf16_t)kvsrc[HDIM + d];
  } else {
    int j = d - HDIM, jj = j & 15;
    float theta = __powf(10000.0f, -(float)jj / 16.0f);
    float ang = (float)s * theta;
    float c = __cosf(ang), sn = __sinf(ang);
    v = (j < 16) ? (kr[j] * c - kr[j + 16] * sn)
                 : (kr[j] * c + kr[j - 16] * sn);
  }
  Kc[(size_t)idx * DQK + d] = (bf16_t)v;
}

// ---------------------------------------------------------------------------
// Flash attention (causal). Block = 4 waves, each wave owns a 16-query tile
// (64 queries per block, same (b,h)). Key blocks of 32. WMMA for QK^T and P*V.
// K tile staged via async copies (ASYNCcnt); V transposed through VALU.
// Output written directly as bf16 [B,S,H*64] = [MTOK, DD] (feeds w_o GEMM).
// ---------------------------------------------------------------------------
__global__ __launch_bounds__(128)
void attn_kernel(const bf16_t* __restrict__ Qc, const bf16_t* __restrict__ Kc,
                 const bf16_t* __restrict__ Vb, bf16_t* __restrict__ Out)
{
  __shared__ bf16_t lK[32 * DQK];       // 32 keys x 96 d        (N-major for B frag)
  __shared__ bf16_t lVt[HDIM * 32];     // 64 d x 32 keys (transposed: N-major over d)
  __shared__ bf16_t lP[4][16 * 32];     // per-wave P staging (A-fragment layout src)

  int blk = blockIdx.x;
  int qblk = blk % (SS / 64);
  int h = (blk / (SS / 64)) % HH;
  int b = blk / ((SS / 64) * HH);

  int t = threadIdx.x;
  int wv = t >> 5, lane = t & 31;
  int hi = lane >> 4, l16 = lane & 15;
  int q0 = qblk * 64 + wv * 16;

  const bf16_t* Qbase = Qc + (size_t)(b * HH + h) * SS * DQK;
  const bf16_t* Kbase = Kc + (size_t)(b * HH + h) * SS * DQK;
  const bf16_t* Vbase = Vb + (size_t)(b * HH + h) * SS * HDIM;

  // Q fragments: 3 chunks of 32 along d, loaded once, kept in registers
  v16bf qf[3];
  {
    const bf16_t* qrow = Qbase + (size_t)(q0 + l16) * DQK;
#pragma unroll
    for (int c = 0; c < 3; ++c)
      qf[c] = mk16(*(const v8bf*)(qrow + c * 32 + hi * 8),
                   *(const v8bf*)(qrow + c * 32 + 16 + hi * 8));
  }

  float mrun[8], lrun[8];
  v8f oacc[4] = {};
#pragma unroll
  for (int r = 0; r < 8; ++r) { mrun[r] = -1e30f; lrun[r] = 0.f; }

  const float scale = 0.10206207262f;   // 1/sqrt(96)
  int nkb = qblk * 2 + 2;               // causal: enough key blocks for this block

  for (int kb = 0; kb < nkb; ++kb) {
    int k0 = kb * 32;
    // stage K tile (32x96) via async copies: key-major == N-major for B frags
    for (int i = t; i < (32 * DQK) / 8; i += 128) {
      int key = i / 12, off = (i % 12) * 8;
      async_load_b128(Kbase + (size_t)(k0 + key) * DQK + off, &lK[key * DQK + off]);
    }
    // stage V tile transposed: lVt[d][key]
    for (int i = t; i < 32 * (HDIM / 8); i += 128) {
      int key = i >> 3, dbase = (i & 7) * 8;
      v8bf vv = *(const v8bf*)(Vbase + (size_t)(k0 + key) * HDIM + dbase);
#pragma unroll
      for (int j = 0; j < 8; ++j) lVt[(dbase + j) * 32 + key] = vv[j];
    }
    wait_async0();
    __syncthreads();

    if (k0 <= q0 + 15) {
      // scores: S[16q x 32k] as two 16x16 C fragments
      v8f sc[2] = {};
#pragma unroll
      for (int kt = 0; kt < 2; ++kt) {
#pragma unroll
        for (int c = 0; c < 3; ++c) {
          int kn = kt * 16 + l16;
          v16bf kfr = mk16(*(const v8bf*)&lK[kn * DQK + c * 32 + hi * 8],
                           *(const v8bf*)&lK[kn * DQK + c * 32 + 16 + hi * 8]);
          sc[kt] = wmma_bf16(qf[c], kfr, sc[kt]);
        }
      }
      // scale + causal mask + online softmax (rows live in 16-lane groups)
#pragma unroll
      for (int r = 0; r < 8; ++r) {
        int q = q0 + hi * 8 + r;
        float v0 = (k0 + l16      <= q) ? sc[0][r] * scale : -1e30f;
        float v1 = (k0 + 16 + l16 <= q) ? sc[1][r] * scale : -1e30f;
        float mx = fmaxf(v0, v1);
#pragma unroll
        for (int off = 1; off < 16; off <<= 1) mx = fmaxf(mx, __shfl_xor(mx, off, 16));
        float mnew = fmaxf(mrun[r], mx);
        float alpha = __expf(mrun[r] - mnew);
        mrun[r] = mnew;
        float p0 = __expf(v0 - mnew);
        float p1 = __expf(v1 - mnew);
        sc[0][r] = p0; sc[1][r] = p1;
        float rs = p0 + p1;
#pragma unroll
        for (int off = 1; off < 16; off <<= 1) rs += __shfl_xor(rs, off, 16);
        lrun[r] = lrun[r] * alpha + rs;
#pragma unroll
        for (int nt = 0; nt < 4; ++nt) oacc[nt][r] *= alpha;
      }
      // stage P (C layout) -> LDS -> reload as A fragment (16x32)
#pragma unroll
      for (int r = 0; r < 8; ++r) {
        int row = hi * 8 + r;
        lP[wv][row * 32 + l16]      = (bf16_t)sc[0][r];
        lP[wv][row * 32 + 16 + l16] = (bf16_t)sc[1][r];
      }
      v16bf pf = mk16(*(const v8bf*)&lP[wv][l16 * 32 + hi * 8],
                      *(const v8bf*)&lP[wv][l16 * 32 + 16 + hi * 8]);
      // O += P @ V  (4 d-tiles of 16)
#pragma unroll
      for (int nt = 0; nt < 4; ++nt) {
        int dn = nt * 16 + l16;
        v16bf vfr = mk16(*(const v8bf*)&lVt[dn * 32 + hi * 8],
                         *(const v8bf*)&lVt[dn * 32 + 16 + hi * 8]);
        oacc[nt] = wmma_bf16(pf, vfr, oacc[nt]);
      }
    }
    __syncthreads();
  }

  // normalize + write bf16 [MTOK, DD] (direct GEMM A-operand)
#pragma unroll
  for (int r = 0; r < 8; ++r) {
    float inv = 1.f / lrun[r];
    int q = q0 + hi * 8 + r;
    size_t base = (size_t)(b * SS + q) * DD + (size_t)h * HDIM;
#pragma unroll
    for (int nt = 0; nt < 4; ++nt)
      Out[base + nt * 16 + l16] = (bf16_t)(oacc[nt][r] * inv);
  }
}

// ---------------------------------------------------------------------------
// SwiGLU gate: g[m,j] = u[m,j] * silu(v[m,j]), bf16 out
// ---------------------------------------------------------------------------
__global__ void glu_kernel(const float* __restrict__ uv, bf16_t* __restrict__ g)
{
  size_t i = (size_t)blockIdx.x * blockDim.x + threadIdx.x;
  if (i < (size_t)MTOK * DFFD) {
    size_t m = i / DFFD, j = i % DFFD;
    float u = uv[m * (2 * DFFD) + j];
    float x = uv[m * (2 * DFFD) + DFFD + j];
    g[i] = (bf16_t)(u * (x / (1.f + __expf(-x))));
  }
}

// ---------------------------------------------------------------------------
// Host orchestration
// ---------------------------------------------------------------------------
static void run_gemm(const float* W, const float* bias, const bf16_t* A,
                     const float* residual, float* C, int M, int N, int K,
                     bf16_t* Wt, hipStream_t stream)
{
  dim3 tg((K + 31) / 32, (N + 31) / 32);
  transpose_bf16_kernel<<<tg, 256, 0, stream>>>(W, Wt, K, N);
  dim3 gg((N + GBN - 1) / GBN, M / GBM);
  gemm_bf16_kernel<<<gg, 256, 0, stream>>>(A, Wt, bias, residual, C, M, N, K);
}

extern "C" void kernel_launch(void* const* d_in, const int* in_sizes, int n_in,
                              void* d_out, int out_size, void* d_ws, size_t ws_size,
                              hipStream_t stream)
{
  (void)in_sizes; (void)n_in; (void)out_size; (void)ws_size;
  const int*   tokens    = (const int*)  d_in[0];
  const float* embed     = (const float*)d_in[1];
  const float* w_kv_down = (const float*)d_in[2];
  const float* b_kv_down = (const float*)d_in[3];
  const float* w_q_down  = (const float*)d_in[4];
  const float* b_q_down  = (const float*)d_in[5];
  const float* w_kv_up   = (const float*)d_in[6];
  const float* b_kv_up   = (const float*)d_in[7];
  const float* w_q_up    = (const float*)d_in[8];
  const float* b_q_up    = (const float*)d_in[9];
  const float* w_o       = (const float*)d_in[10];
  const float* b_o       = (const float*)d_in[11];
  const float* kv_norm_w = (const float*)d_in[12];
  const float* q_norm_w  = (const float*)d_in[13];
  const float* norm1_w   = (const float*)d_in[14];
  const float* norm2_w   = (const float*)d_in[15];
  const float* w_in      = (const float*)d_in[16];
  const float* b_in      = (const float*)d_in[17];
  const float* w_out     = (const float*)d_in[18];
  const float* b_out     = (const float*)d_in[19];
  const float* norm_f_w  = (const float*)d_in[20];
  const float* w_head    = (const float*)d_in[21];
  const float* b_head    = (const float*)d_in[22];
  float* out = (float*)d_out;

  char* p = (char*)d_ws;
  auto alloc = [&](size_t bytes) -> void* {
    void* r = (void*)p; p += (bytes + 255) & ~(size_t)255; return r;
  };
  bf16_t* Wt    = (bf16_t*)alloc((size_t)VOCAB * DD * 2);          // 65.5 MB (max N*K)
  float*  h     = (float*) alloc((size_t)MTOK * DD * 4);
  bf16_t* xn    = (bf16_t*)alloc((size_t)MTOK * DD * 2);
  float*  ckv   = (float*) alloc((size_t)MTOK * (CKVD + RRD) * 4);
  bf16_t* ckvn  = (bf16_t*)alloc((size_t)MTOK * CKVD * 2);
  float*  kvup  = (float*) alloc((size_t)MTOK * (2 * HDIM * HH) * 4);
  float*  qd    = (float*) alloc((size_t)MTOK * CQD * 4);
  bf16_t* qdn   = (bf16_t*)alloc((size_t)MTOK * CQD * 2);
  float*  qup   = (float*) alloc((size_t)MTOK * (DQK * HH) * 4);
  bf16_t* Qc    = (bf16_t*)alloc((size_t)BB * HH * SS * DQK * 2);
  bf16_t* Kc    = (bf16_t*)alloc((size_t)BB * HH * SS * DQK * 2);
  bf16_t* Vb    = (bf16_t*)alloc((size_t)BB * HH * SS * HDIM * 2);
  bf16_t* attnb = (bf16_t*)alloc((size_t)MTOK * DD * 2);
  float*  uv    = (float*) alloc((size_t)MTOK * 2 * DFFD * 4);
  bf16_t* gb    = (bf16_t*)alloc((size_t)MTOK * DFFD * 2);

  embed_kernel<<<MTOK, 256, 0, stream>>>(tokens, embed, h);

  for (int i = 0; i < NLAYERS; ++i) {
    // xn = rmsnorm(h, norm1)
    rmsnorm_kernel<<<MTOK, 256, 0, stream>>>(h, DD, norm1_w + (size_t)i * DD, xn, DD);
    // ckv = xn @ w_kv_down + b
    run_gemm(w_kv_down + (size_t)i * DD * (CKVD + RRD), b_kv_down + (size_t)i * (CKVD + RRD),
             xn, nullptr, ckv, MTOK, CKVD + RRD, DD, Wt, stream);
    // kv = rmsnorm(ckv[:, :256]) @ w_kv_up + b
    rmsnorm_kernel<<<MTOK, 256, 0, stream>>>(ckv, CKVD + RRD, kv_norm_w + (size_t)i * CKVD, ckvn, CKVD);
    run_gemm(w_kv_up + (size_t)i * CKVD * (2 * HDIM * HH), b_kv_up + (size_t)i * (2 * HDIM * HH),
             ckvn, nullptr, kvup, MTOK, 2 * HDIM * HH, CKVD, Wt, stream);
    // q path
    run_gemm(w_q_down + (size_t)i * DD * CQD, b_q_down + (size_t)i * CQD,
             xn, nullptr, qd, MTOK, CQD, DD, Wt, stream);
    rmsnorm_kernel<<<MTOK, 256, 0, stream>>>(qd, CQD, q_norm_w + (size_t)i * CQD, qdn, CQD);
    run_gemm(w_q_up + (size_t)i * CQD * (DQK * HH), b_q_up + (size_t)i * (DQK * HH),
             qdn, nullptr, qup, MTOK, DQK * HH, CQD, Wt, stream);
    // rope packs + attention
    build_q_kernel <<<BB * HH * SS, DQK, 0, stream>>>(qup, Qc);
    build_kv_kernel<<<BB * HH * SS, DQK, 0, stream>>>(kvup, ckv, Kc, Vb);
    attn_kernel<<<BB * HH * (SS / 64), 128, 0, stream>>>(Qc, Kc, Vb, attnb);
    // h = h + attn @ w_o + b_o
    run_gemm(w_o + (size_t)i * DD * DD, b_o + (size_t)i * DD, attnb, h, h, MTOK, DD, DD, Wt, stream);
    // MLP
    rmsnorm_kernel<<<MTOK, 256, 0, stream>>>(h, DD, norm2_w + (size_t)i * DD, xn, DD);
    run_gemm(w_in + (size_t)i * DD * 2 * DFFD, b_in + (size_t)i * 2 * DFFD,
             xn, nullptr, uv, MTOK, 2 * DFFD, DD, Wt, stream);
    glu_kernel<<<((size_t)MTOK * DFFD + 255) / 256, 256, 0, stream>>>(uv, gb);
    run_gemm(w_out + (size_t)i * DFFD * DD, b_out + (size_t)i * DD, gb, h, h, MTOK, DD, DFFD, Wt, stream);
  }

  // head
  rmsnorm_kernel<<<MTOK, 256, 0, stream>>>(h, DD, norm_f_w, xn, DD);
  run_gemm(w_head, b_head, xn, nullptr, out, MTOK, VOCAB, DD, Wt, stream);
}
